// eye_16750372454700
// MI455X (gfx1250) — compile-verified, matrix operand order checked by reading, not run on hardware
//
#include <hip/hip_runtime.h>
#include <hip/hip_bf16.h>

// ---------- types for WMMA ----------
typedef __attribute__((ext_vector_type(16))) __bf16 v16bf;
typedef __attribute__((ext_vector_type(8)))  float  v8f;
typedef __attribute__((ext_vector_type(8)))  int    v8i;

union BF16x16 { v8i u; v16bf v; };

// ---------- helpers ----------
__device__ __forceinline__ unsigned short f2bf(float f) {
    unsigned int u = __float_as_uint(f);
    u += 0x7FFFu + ((u >> 16) & 1u);          // round-to-nearest-even
    return (unsigned short)(u >> 16);
}
__device__ __forceinline__ float bf2f(unsigned short h) {
    return __uint_as_float(((unsigned int)h) << 16);
}
__device__ __forceinline__ float sigmoidf(float x) {
    return 1.0f / (1.0f + __expf(-x));
}

// K offset inside a 32-wide K-slab for 16-bit A/B fragments (ISA 7.12.2):
// VGPR v: K = 2*(v&3) + 16*(v>>2) + 8*laneHi
__device__ __forceinline__ int frag_kk(int v, int hi) {
    return ((v & 3) << 1) + ((v >> 2) << 4) + (hi << 3);
}

// ---------- 1) transpose + convert f32 [K][N] -> bf16 [N][Kp] (zero pad) ----------
__global__ void transpose_bf16_kernel(const float* __restrict__ src,
                                      unsigned short* __restrict__ dst,
                                      int K, int N, int Kp) {
    int idx = blockIdx.x * 256 + threadIdx.x;
    if (idx >= N * Kp) return;
    int n = idx / Kp, k = idx % Kp;
    dst[idx] = (k < K) ? f2bf(src[(size_t)k * N + n]) : (unsigned short)0;
}

// ---------- 2) segment reduce: hidden [64,512,768], mask -> X1 bf16 [3200,800] cols 0..767 ----------
__global__ void segment_reduce_kernel(const float* __restrict__ hidden,
                                      const int* __restrict__ mask,
                                      unsigned short* __restrict__ X1) {
    int b = blockIdx.x;       // 0..63
    int ch = blockIdx.y;      // 0..11 (chunks of 64 dims)
    __shared__ float bins[50 * 64];
    int tid = threadIdx.x;    // 256
    for (int i = tid; i < 50 * 64; i += 256) bins[i] = 0.0f;
    __syncthreads();
    int d = tid & 63;
    int ts = tid >> 6;        // 0..3
    for (int t0 = 0; t0 < 512; t0 += 4) {
        int t = t0 + ts;
        int m = mask[b * 512 + t];
        float v = hidden[((size_t)(b * 512 + t)) * 768 + ch * 64 + d];
        atomicAdd(&bins[m * 64 + d], v);
    }
    __syncthreads();
    for (int i = tid; i < 50 * 64; i += 256) {
        int m = i >> 6, dd = i & 63;
        X1[((size_t)(b * 50 + m)) * 800 + ch * 64 + dd] = f2bf(bins[i] * (1.0f / 50.0f));
    }
}

// ---------- 3) concat lstm_input into cols 768..775, zero-pad 776..799 ----------
__global__ void concat_tail_kernel(const float* __restrict__ lstm_input,
                                   unsigned short* __restrict__ X1) {
    int row = blockIdx.x;     // 0..3199
    int j = threadIdx.x;      // 32 threads
    if (j < 8)
        X1[(size_t)row * 800 + 768 + j] = f2bf(lstm_input[row * 8 + j]);
    else
        X1[(size_t)row * 800 + 776 + (j - 8)] = 0;
}

// ---------- 4) WMMA GEMM: C[M,N] f32 = A[M,Kp]bf16 @ Bt[N,Kp]bf16^T + bias ----------
// grid: (M/16, N/512), block 256 = 8 waves; each wave computes a 16x64 strip
// (4 accumulators) so the A-fragment is loaded once per K-slab and reused 4x.
__global__ void gemm_bias_wmma_kernel(const unsigned short* __restrict__ A,
                                      const unsigned short* __restrict__ Bt,
                                      const float* __restrict__ bias,
                                      float* __restrict__ C, int N, int Kp) {
    int tid = threadIdx.x;
    int lane = tid & 31, wave = tid >> 5;
    int hi = (lane >> 4) & 1, ln = lane & 15;
    int m0 = blockIdx.x * 16;
    int n0 = blockIdx.y * 512 + wave * 64;
    v8f acc[4];
    #pragma unroll
    for (int nt = 0; nt < 4; ++nt) {
        float bv = bias[n0 + nt * 16 + ln];
        #pragma unroll
        for (int i = 0; i < 8; ++i) acc[nt][i] = bv;
    }
    const unsigned short* arow = A + (size_t)(m0 + ln) * Kp;
    const unsigned short* brow = Bt + (size_t)(n0 + ln) * Kp;
    for (int k0 = 0; k0 < Kp; k0 += 32) {
        BF16x16 af;
        #pragma unroll
        for (int v = 0; v < 8; ++v)
            af.u[v] = *(const int*)(arow + k0 + frag_kk(v, hi));
        #pragma unroll
        for (int nt = 0; nt < 4; ++nt) {
            BF16x16 bf;
            const unsigned short* bp = brow + (size_t)nt * 16 * Kp + k0;
            #pragma unroll
            for (int v = 0; v < 8; ++v)
                bf.u[v] = *(const int*)(bp + frag_kk(v, hi));
            acc[nt] = __builtin_amdgcn_wmma_f32_16x16x32_bf16(false, af.v, false, bf.v,
                                                              (short)0, acc[nt], false, false);
        }
    }
    #pragma unroll
    for (int nt = 0; nt < 4; ++nt) {
        #pragma unroll
        for (int i = 0; i < 8; ++i) {
            int Mi = i + (hi << 3);
            C[(size_t)(m0 + Mi) * N + n0 + nt * 16 + ln] = acc[nt][i];
        }
    }
}

// ---------- 5) LSTM time scan with WMMA recurrence ----------
// grid (B/16, 2 dirs), block 512 = 16 waves. h,c live in LDS across 50 steps.
// A-fragment (h) loaded once per K-slab, reused across NT gate tiles; next
// timestep's xz slab is prefetched (global_prefetch_b8) behind the WMMA work.
template <int U>
__global__ void lstm_scan_kernel(const float* __restrict__ xzF,
                                 const float* __restrict__ xzB,
                                 const unsigned short* __restrict__ WrF,
                                 const unsigned short* __restrict__ WrB,
                                 unsigned short* __restrict__ seq_out,
                                 int outStride, int offF, int offB) {
    constexpr int NG = 4 * U;          // gate columns
    constexpr int NT = NG / 256;       // n-tiles per wave (16 waves x 16 cols)
    constexpr int T = 50;
    __shared__ unsigned short zbuf[16 * NG];   // gate pre-activations (bf16)
    __shared__ float          cbuf[16 * U];    // cell state (f32)
    __shared__ unsigned short hbuf[16 * U];    // hidden state (bf16) = WMMA A
    int dir = blockIdx.y;
    const float* xz = dir ? xzB : xzF;
    const unsigned short* Wr = dir ? WrB : WrF;
    int outOff = dir ? offB : offF;
    int tid = threadIdx.x;
    int lane = tid & 31, wave = tid >> 5;
    int hi = (lane >> 4) & 1, ln = lane & 15;
    int b0 = blockIdx.x * 16;
    for (int i = tid; i < 16 * U; i += 512) { cbuf[i] = 0.0f; hbuf[i] = 0; }
    __syncthreads();
    for (int s = 0; s < T; ++s) {
        int t = dir ? (T - 1 - s) : s;   // bwd consumes/stores reversed time
        // prefetch next step's xz slab (16 rows x NG f32) into cache
        if (s + 1 < T) {
            int tn = dir ? (T - 2 - s) : (s + 1);
            int f = tid * (NG / 32);     // flat element index, NG/32-float (NG/8 B) stride
            int pm = f / NG, pc = f % NG;
            __builtin_prefetch(&xz[((size_t)(b0 + pm) * T + tn) * NG + pc], 0, 0);
        }
        // z = xz[:,t,:] + h @ Wr  (accumulators seeded from xz)
        v8f acc[NT];
        #pragma unroll
        for (int nt = 0; nt < NT; ++nt) {
            int n0 = (wave * NT + nt) * 16;
            #pragma unroll
            for (int i = 0; i < 8; ++i) {
                int Mi = i + (hi << 3);
                acc[nt][i] = xz[((size_t)(b0 + Mi) * T + t) * NG + n0 + ln];
            }
        }
        const unsigned short* arow = hbuf + ln * U;
        #pragma unroll
        for (int k0 = 0; k0 < U; k0 += 32) {
            BF16x16 af;
            #pragma unroll
            for (int v = 0; v < 8; ++v)
                af.u[v] = *(const int*)(arow + k0 + frag_kk(v, hi));
            #pragma unroll
            for (int nt = 0; nt < NT; ++nt) {
                int n0 = (wave * NT + nt) * 16;
                const unsigned short* bp = Wr + (size_t)(n0 + ln) * U + k0;
                BF16x16 bf;
                #pragma unroll
                for (int v = 0; v < 8; ++v)
                    bf.u[v] = *(const int*)(bp + frag_kk(v, hi));
                acc[nt] = __builtin_amdgcn_wmma_f32_16x16x32_bf16(false, af.v, false, bf.v,
                                                                  (short)0, acc[nt], false, false);
            }
        }
        #pragma unroll
        for (int nt = 0; nt < NT; ++nt) {
            int n0 = (wave * NT + nt) * 16;
            #pragma unroll
            for (int i = 0; i < 8; ++i) {
                int Mi = i + (hi << 3);
                zbuf[Mi * NG + n0 + ln] = f2bf(acc[nt][i]);
            }
        }
        __syncthreads();
        // gates -> c,h update (Keras order i,f,g,o)
        for (int idx = tid; idx < 16 * U; idx += 512) {
            int m = idx / U, j = idx % U;
            float iv = sigmoidf(bf2f(zbuf[m * NG + j]));
            float fv = sigmoidf(bf2f(zbuf[m * NG + U + j]));
            float gv = tanhf(bf2f(zbuf[m * NG + 2 * U + j]));
            float ov = sigmoidf(bf2f(zbuf[m * NG + 3 * U + j]));
            float c = fv * cbuf[idx] + iv * gv;
            cbuf[idx] = c;
            unsigned short hb = f2bf(ov * tanhf(c));
            hbuf[idx] = hb;
            seq_out[((size_t)(b0 + m) * T + t) * outStride + outOff + j] = hb;
        }
        __syncthreads();
    }
}

// ---------- 6) dense head: [256]->128->64->32->1 ----------
__global__ void dense_head_kernel(const unsigned short* __restrict__ H2,
                                  const float* d1w, const float* d1b,
                                  const float* d3w, const float* d3b,
                                  const float* d5w, const float* d5b,
                                  const float* d7w, const float* d7b,
                                  float* __restrict__ out) {
    int b = blockIdx.x, tid = threadIdx.x;  // 128 threads
    __shared__ float x[256], y1[128], y2[64], y3[32];
    // fwd last state at t=49 (cols 0..127); bwd last state at t=0 (cols 128..255)
    x[tid]       = bf2f(H2[((size_t)b * 50 + 49) * 256 + tid]);
    x[128 + tid] = bf2f(H2[((size_t)b * 50 + 0) * 256 + 128 + tid]);
    __syncthreads();
    { float s = d1b[tid];
      for (int k = 0; k < 256; ++k) s += x[k] * d1w[k * 128 + tid];
      y1[tid] = fmaxf(s, 0.0f); }
    __syncthreads();
    if (tid < 64) { float s = d3b[tid];
        for (int k = 0; k < 128; ++k) s += y1[k] * d3w[k * 64 + tid];
        y2[tid] = fmaxf(s, 0.0f); }
    __syncthreads();
    if (tid < 32) { float s = d5b[tid];
        for (int k = 0; k < 64; ++k) s += y2[k] * d5w[k * 32 + tid];
        y3[tid] = fmaxf(s, 0.0f); }
    __syncthreads();
    if (tid == 0) { float s = d7b[0];
        for (int k = 0; k < 32; ++k) s += y3[k] * d7w[k];
        out[b] = 1.0f / (1.0f + __expf(-s)); }
}

// ---------- host ----------
extern "C" void kernel_launch(void* const* d_in, const int* in_sizes, int n_in,
                              void* d_out, int out_size, void* d_ws, size_t ws_size,
                              hipStream_t stream) {
    (void)in_sizes; (void)n_in; (void)out_size; (void)ws_size;
    const float* hidden     = (const float*)d_in[0];
    const int*   mask       = (const int*)d_in[1];
    const float* lstm_input = (const float*)d_in[2];
    const float* l1f_k = (const float*)d_in[4];
    const float* l1f_r = (const float*)d_in[5];
    const float* l1f_b = (const float*)d_in[6];
    const float* l1b_k = (const float*)d_in[7];
    const float* l1b_r = (const float*)d_in[8];
    const float* l1b_b = (const float*)d_in[9];
    const float* l2f_k = (const float*)d_in[10];
    const float* l2f_r = (const float*)d_in[11];
    const float* l2f_b = (const float*)d_in[12];
    const float* l2b_k = (const float*)d_in[13];
    const float* l2b_r = (const float*)d_in[14];
    const float* l2b_b = (const float*)d_in[15];
    const float* d1w = (const float*)d_in[16];
    const float* d1b = (const float*)d_in[17];
    const float* d3w = (const float*)d_in[18];
    const float* d3b = (const float*)d_in[19];
    const float* d5w = (const float*)d_in[20];
    const float* d5b = (const float*)d_in[21];
    const float* d7w = (const float*)d_in[22];
    const float* d7b = (const float*)d_in[23];
    float* out = (float*)d_out;

    char* ws = (char*)d_ws;
    size_t off = 0;
    auto alloc = [&](size_t bytes) -> char* {
        char* p = ws + off;
        off = (off + bytes + 255) & ~(size_t)255;
        return p;
    };
    unsigned short* w1f = (unsigned short*)alloc(1024 * 800 * 2);
    unsigned short* w1b = (unsigned short*)alloc(1024 * 800 * 2);
    unsigned short* r1f = (unsigned short*)alloc(1024 * 256 * 2);
    unsigned short* r1b = (unsigned short*)alloc(1024 * 256 * 2);
    unsigned short* w2f = (unsigned short*)alloc(512 * 512 * 2);
    unsigned short* w2b = (unsigned short*)alloc(512 * 512 * 2);
    unsigned short* r2f = (unsigned short*)alloc(512 * 128 * 2);
    unsigned short* r2b = (unsigned short*)alloc(512 * 128 * 2);
    unsigned short* X1  = (unsigned short*)alloc((size_t)3200 * 800 * 2);
    unsigned short* A2  = (unsigned short*)alloc((size_t)3200 * 512 * 2);
    unsigned short* H2  = (unsigned short*)alloc((size_t)3200 * 256 * 2);
    float* xz1f = (float*)alloc((size_t)3200 * 1024 * 4);
    float* xz1b = (float*)alloc((size_t)3200 * 1024 * 4);
    float* xz2f = (float*)alloc((size_t)3200 * 512 * 4);
    float* xz2b = (float*)alloc((size_t)3200 * 512 * 4);

    // weight conversion (transpose, pad K, bf16)
    transpose_bf16_kernel<<<(1024 * 800 + 255) / 256, 256, 0, stream>>>(l1f_k, w1f, 776, 1024, 800);
    transpose_bf16_kernel<<<(1024 * 800 + 255) / 256, 256, 0, stream>>>(l1b_k, w1b, 776, 1024, 800);
    transpose_bf16_kernel<<<(1024 * 256 + 255) / 256, 256, 0, stream>>>(l1f_r, r1f, 256, 1024, 256);
    transpose_bf16_kernel<<<(1024 * 256 + 255) / 256, 256, 0, stream>>>(l1b_r, r1b, 256, 1024, 256);
    transpose_bf16_kernel<<<(512 * 512 + 255) / 256, 256, 0, stream>>>(l2f_k, w2f, 512, 512, 512);
    transpose_bf16_kernel<<<(512 * 512 + 255) / 256, 256, 0, stream>>>(l2b_k, w2b, 512, 512, 512);
    transpose_bf16_kernel<<<(512 * 128 + 255) / 256, 256, 0, stream>>>(l2f_r, r2f, 128, 512, 128);
    transpose_bf16_kernel<<<(512 * 128 + 255) / 256, 256, 0, stream>>>(l2b_r, r2b, 128, 512, 128);

    // segment reduce + concat
    segment_reduce_kernel<<<dim3(64, 12), 256, 0, stream>>>(hidden, mask, X1);
    concat_tail_kernel<<<3200, 32, 0, stream>>>(lstm_input, X1);

    // layer-1 input projections  (N=1024 -> grid.y = 2)
    gemm_bias_wmma_kernel<<<dim3(200, 2), 256, 0, stream>>>(X1, w1f, l1f_b, xz1f, 1024, 800);
    gemm_bias_wmma_kernel<<<dim3(200, 2), 256, 0, stream>>>(X1, w1b, l1b_b, xz1b, 1024, 800);

    // layer-1 scan (both directions), writes bf16 sequence into A2 [3200,512]
    lstm_scan_kernel<256><<<dim3(4, 2), 512, 0, stream>>>(xz1f, xz1b, r1f, r1b, A2, 512, 0, 256);

    // layer-2 input projections  (N=512 -> grid.y = 1)
    gemm_bias_wmma_kernel<<<dim3(200, 1), 256, 0, stream>>>(A2, w2f, l2f_b, xz2f, 512, 512);
    gemm_bias_wmma_kernel<<<dim3(200, 1), 256, 0, stream>>>(A2, w2b, l2b_b, xz2b, 512, 512);

    // layer-2 scan, writes H2 [3200,256]
    lstm_scan_kernel<128><<<dim3(4, 2), 512, 0, stream>>>(xz2f, xz2b, r2f, r2b, H2, 256, 0, 128);

    // dense head
    dense_head_kernel<<<64, 128, 0, stream>>>(H2, d1w, d1b, d3w, d3b, d5w, d5b, d7w, d7b, out);
}
